// QMCrossAttention_8684423872626
// MI455X (gfx1250) — matrix operand order repaired and checked by reading, not compiled, not run on hardware
//
#include <hip/hip_runtime.h>
#include <hip/hip_bf16.h>
#include <cmath>

// ---------------------------------------------------------------------------
// QMCrossAttention for MI455X (gfx1250, wave32, WMMA)
// v3: score tile computed TRANSPOSED (S^T = K Q^T) so softmax-over-keys is an
// in-lane VALU tree + one cross-half shuffle (2 bpermutes/block instead of 64).
// K frags double-buffered, V frags hoisted, GEMM software-pipelined.
// ---------------------------------------------------------------------------

typedef __attribute__((ext_vector_type(16))) _Float16 v16h;
typedef __attribute__((ext_vector_type(8)))  _Float16 v8h;
typedef __attribute__((ext_vector_type(8)))  float    v8f;

union ABFrag { v16h v; v8h h[2]; };

#define WMMA_F16(a, b, c) \
  __builtin_amdgcn_wmma_f32_16x16x32_f16(false, (a), false, (b), (short)0, (c), false, false)

static constexpr int B_   = 8;
static constexpr int N_   = 2048;
static constexpr int M_   = 4096;
static constexpr int HID  = 256;
static constexpr int H_   = 4;

// -------------------------------------------------------------- prep kernels
__global__ __launch_bounds__(256) void cvt_f32_to_f16(const float* __restrict__ src,
                                                      _Float16* __restrict__ dst, int n) {
  for (int i = blockIdx.x * blockDim.x + threadIdx.x; i < n; i += gridDim.x * blockDim.x)
    dst[i] = (_Float16)src[i];
}

// W is [K=256][N=256] row-major f32; produce Wt[N][K] f16 so WMMA B-fragments
// (two 8-K runs per lane) are contiguous b128 loads.
__global__ __launch_bounds__(256) void transpose256(const float* __restrict__ W,
                                                    _Float16* __restrict__ Wt) {
  int k = blockIdx.x * 16 + threadIdx.x;
  int n = blockIdx.y * 16 + threadIdx.y;
  Wt[n * 256 + k] = (_Float16)W[k * 256 + n];
}

// ---------------------------------------------------------------- GEMM (proj)
// Y = X(f16, Rx256 row-major) @ W(via Wt f16 [N][K]) + bias(f32)
// Each wave computes a 16x32 tile (two C frags sharing one A frag),
// K-loop software-pipelined with one-step lookahead.
// MODE 0: f16 out, head-split   ((b*4 + c/64)*RPB + r%RPB)*64 + c%64   (Q, K)
// MODE 1: f16 out, head-split transposed ((b*4+c/64)*64 + c%64)*RPB + r%RPB (V^T)
// MODE 2: f32 out, row-major r*256 + c                                  (final)
template <int MODE>
__global__ __launch_bounds__(128) void gemm_proj(const _Float16* __restrict__ X,
                                                 const _Float16* __restrict__ Wt,
                                                 const float* __restrict__ bias,
                                                 void* __restrict__ out,
                                                 int rowsPerBatch) {
  const int lane  = threadIdx.x & 31;
  const int wave  = threadIdx.x >> 5;
  const int rowTile = blockIdx.x >> 1;
  const int colPair = (blockIdx.x & 1) * 4 + wave;   // 0..7 -> 32 cols each
  const int r0 = rowTile * 16;
  const int c0 = colPair * 32;
  const int lhalf = lane >> 4;
  const int lm    = lane & 15;

  const _Float16* xrow  = X  + (size_t)(r0 + lm) * 256;
  const _Float16* wrow0 = Wt + (size_t)(c0 + lm) * 256;
  const _Float16* wrow1 = Wt + (size_t)(c0 + 16 + lm) * 256;

  auto loadA = [&](int kk, ABFrag& a) {
    a.h[0] = *(const v8h*)(xrow + kk + lhalf * 8);
    a.h[1] = *(const v8h*)(xrow + kk + 16 + lhalf * 8);
  };
  auto loadB = [&](const _Float16* wr, int kk, ABFrag& b) {
    b.h[0] = *(const v8h*)(wr + kk + lhalf * 8);
    b.h[1] = *(const v8h*)(wr + kk + 16 + lhalf * 8);
  };

  v8f acc0 = {}, acc1 = {};
  ABFrag a, b0, b1;
  loadA(0, a); loadB(wrow0, 0, b0); loadB(wrow1, 0, b1);
#pragma unroll
  for (int k8 = 0; k8 < 8; ++k8) {
    const int kk = k8 * 32;
    ABFrag an, bn0, bn1;
    if (k8 < 7) {
      loadA(kk + 32, an); loadB(wrow0, kk + 32, bn0); loadB(wrow1, kk + 32, bn1);
    }
    acc0 = WMMA_F16(a.v, b0.v, acc0);
    acc1 = WMMA_F16(a.v, b1.v, acc1);
    if (k8 < 7) { a = an; b0 = bn0; b1 = bn1; }
  }

  const float bv0 = bias[c0 + lm];
  const float bv1 = bias[c0 + 16 + lm];
#pragma unroll
  for (int t = 0; t < 2; ++t) {
    const v8f& acc = t ? acc1 : acc0;
    const float bv = t ? bv1 : bv0;
#pragma unroll
    for (int j = 0; j < 8; ++j) {
      const int r = r0 + j + lhalf * 8;
      const int c = c0 + t * 16 + lm;
      const float val = acc[j] + bv;
      if (MODE == 0) {
        const int b_ = r / rowsPerBatch, rb = r % rowsPerBatch;
        ((_Float16*)out)[((size_t)(b_ * 4 + (c >> 6)) * rowsPerBatch + rb) * 64 + (c & 63)] =
            (_Float16)val;
      } else if (MODE == 1) {
        const int b_ = r / rowsPerBatch, rb = r % rowsPerBatch;
        ((_Float16*)out)[((size_t)(b_ * 4 + (c >> 6)) * 64 + (c & 63)) * (size_t)rowsPerBatch + rb] =
            (_Float16)val;
      } else {
        ((float*)out)[(size_t)r * 256 + c] = val;
      }
    }
  }
}

// ---------------------------------------------------------- fused attention
// One wave owns a 16-row Q tile of one (b,h); loops keys in blocks of 32.
// S^T = K Q^T so each lane holds one q-row's scores for the whole block:
// softmax reduction = in-lane tree + one shfl_xor(16). O accumulates in
// normal orientation; alpha broadcast back via 8 bpermutes/block.
__global__ __launch_bounds__(128) void attention_kernel(
    const _Float16* __restrict__ Q,    // [B*H][N][64]
    const _Float16* __restrict__ K,    // [B*H][M][64]
    const _Float16* __restrict__ Vt,   // [B*H][64][M]
    _Float16* __restrict__ O)          // [B*N][256]
{
  __shared__ __align__(16) _Float16 plds[4 * 16 * 40];  // per-wave 16x32, stride 40

  const int lane  = threadIdx.x & 31;
  const int wave  = threadIdx.x >> 5;
  const int bh    = blockIdx.x >> 5;   // 0..31 = b*4 + h
  const int chunk = blockIdx.x & 31;   // 0..31 -> 64 q rows each
  const int b_ = bh >> 2, h_ = bh & 3;
  const int n0 = chunk * 64 + wave * 16;
  const int lhalf = lane >> 4;
  const int lm    = lane & 15;

  // Q tile as two B-fragments (Q^T: lane = q-row, two 8-d runs per lane)
  const _Float16* qbase = Q + ((size_t)bh * N_ + n0 + lm) * 64;
  ABFrag qb[2];
#pragma unroll
  for (int dc = 0; dc < 2; ++dc) {
    qb[dc].h[0] = *(const v8h*)(qbase + dc * 32 + lhalf * 8);
    qb[dc].h[1] = *(const v8h*)(qbase + dc * 32 + 16 + lhalf * 8);
  }

  float mrow = -__builtin_inff();   // per-lane: stats of q-row n0 + lm
  float lrow = 0.0f;
  v8f oacc[4] = {};

  _Float16* pw = plds + wave * 16 * 40;
  const _Float16* kbh = K  + (size_t)bh * M_ * 64;
  const _Float16* vbh = Vt + (size_t)bh * 64 * M_;
  const float scale_log2e = 0.18033688011112042f;  // (1/sqrt(64)) * log2(e)

  // K block as A-fragments: lane = key row, d chunks split by lane half
  auto loadK = [&](int mt, ABFrag kf[2][2]) {
#pragma unroll
    for (int s = 0; s < 2; ++s) {
      const _Float16* kr = kbh + (size_t)(mt + s * 16 + lm) * 64;
#pragma unroll
      for (int dc = 0; dc < 2; ++dc) {
        kf[s][dc].h[0] = *(const v8h*)(kr + dc * 32 + lhalf * 8);
        kf[s][dc].h[1] = *(const v8h*)(kr + dc * 32 + 16 + lhalf * 8);
      }
    }
  };

  ABFrag kf[2][2];
  loadK(0, kf);  // preload first K block

#pragma unroll 2
  for (int mt = 0; mt < M_; mt += 32) {
    // ---- issue V loads for this block (consumed only after softmax)
    ABFrag vf[4];
#pragma unroll
    for (int f = 0; f < 4; ++f) {
      const _Float16* vr = vbh + (size_t)(f * 16 + lm) * M_ + mt + lhalf * 8;
      vf[f].h[0] = *(const v8h*)(vr);
      vf[f].h[1] = *(const v8h*)(vr + 16);
    }
    // ---- issue next block's K loads (double buffer)
    ABFrag kn[2][2];
    const bool more = (mt + 32 < M_);
    if (more) loadK(mt + 32, kn);

    // ---- S^T = K Q^T : tile s has M-dim = keys s*16.., N-dim = q-rows
    v8f st[2] = {};
#pragma unroll
    for (int s = 0; s < 2; ++s)
#pragma unroll
      for (int dc = 0; dc < 2; ++dc)
        st[s] = WMMA_F16(kf[s][dc].v, qb[dc].v, st[s]);

    // ---- softmax over keys: all 32 scores of q-row (n0+lm) live in this
    //      lane pair (this half: keys j/j+8 of each subtile)
    float s2[2][8];
#pragma unroll
    for (int s = 0; s < 2; ++s)
#pragma unroll
      for (int j = 0; j < 8; ++j) s2[s][j] = st[s][j] * scale_log2e;

    float tmax = fmaxf(
        fmaxf(fmaxf(fmaxf(s2[0][0], s2[0][1]), fmaxf(s2[0][2], s2[0][3])),
              fmaxf(fmaxf(s2[0][4], s2[0][5]), fmaxf(s2[0][6], s2[0][7]))),
        fmaxf(fmaxf(fmaxf(s2[1][0], s2[1][1]), fmaxf(s2[1][2], s2[1][3])),
              fmaxf(fmaxf(s2[1][4], s2[1][5]), fmaxf(s2[1][6], s2[1][7]))));
    tmax = fmaxf(tmax, __shfl_xor(tmax, 16, 32));  // combine lane halves

    const float mnew  = fmaxf(mrow, tmax);
    const float alpha = exp2f(mrow - mnew);
    mrow = mnew;

    float p[2][8];
#pragma unroll
    for (int s = 0; s < 2; ++s)
#pragma unroll
      for (int j = 0; j < 8; ++j) p[s][j] = exp2f(s2[s][j] - mnew);

    float rs = ((p[0][0] + p[0][1]) + (p[0][2] + p[0][3])) +
               ((p[0][4] + p[0][5]) + (p[0][6] + p[0][7])) +
               ((p[1][0] + p[1][1]) + (p[1][2] + p[1][3])) +
               ((p[1][4] + p[1][5]) + (p[1][6] + p[1][7]));
    rs += __shfl_xor(rs, 16, 32);
    lrow = lrow * alpha + rs;

    // ---- stage P^T (C-layout of S^T) into LDS as P[n][m] row-major f16
#pragma unroll
    for (int s = 0; s < 2; ++s)
#pragma unroll
      for (int j = 0; j < 8; ++j)
        pw[lm * 40 + s * 16 + j + lhalf * 8] = (_Float16)p[s][j];

    // ---- rescale O accumulator: broadcast alpha from lane (row) to all
#pragma unroll
    for (int j = 0; j < 8; ++j) {
      const float aj = __shfl(alpha, j + 8 * lhalf, 32);
#pragma unroll
      for (int f = 0; f < 4; ++f) oacc[f][j] *= aj;
    }

    // ---- reload P as a 16x32 A-fragment (DS ops are in-order per wave)
    ABFrag pf;
    const _Float16* pr = pw + lm * 40;
    pf.h[0] = *(const v8h*)(pr + lhalf * 8);
    pf.h[1] = *(const v8h*)(pr + 16 + lhalf * 8);

    // ---- O += P V : V frags were issued at top of iteration
#pragma unroll
    for (int f = 0; f < 4; ++f)
      oacc[f] = WMMA_F16(pf.v, vf[f].v, oacc[f]);

    // ---- rotate K double buffer
    if (more) {
#pragma unroll
      for (int s = 0; s < 2; ++s)
#pragma unroll
        for (int dc = 0; dc < 2; ++dc) kf[s][dc] = kn[s][dc];
    }
  }

  // ---- normalize (broadcast 1/l per row) and store O[b][n][h*64+d] as f16
  const float linv = 1.0f / lrow;
#pragma unroll
  for (int j = 0; j < 8; ++j) {
    const float lj = __shfl(linv, j + 8 * lhalf, 32);
    const int n = n0 + j + lhalf * 8;
#pragma unroll
    for (int f = 0; f < 4; ++f) {
      const int d = f * 16 + lm;
      O[((size_t)b_ * N_ + n) * 256 + h_ * 64 + d] = (_Float16)(oacc[f][j] * lj);
    }
  }
}

// ----------------------------------------------------------------- launcher
extern "C" void kernel_launch(void* const* d_in, const int* in_sizes, int n_in,
                              void* d_out, int out_size, void* d_ws, size_t ws_size,
                              hipStream_t stream) {
  const float* nuc = (const float*)d_in[0];
  const float* ele = (const float*)d_in[1];
  const float* Wq  = (const float*)d_in[2];
  const float* bq  = (const float*)d_in[3];
  const float* Wk  = (const float*)d_in[4];
  const float* bk  = (const float*)d_in[5];
  const float* Wv  = (const float*)d_in[6];
  const float* bv  = (const float*)d_in[7];
  const float* Wo  = (const float*)d_in[8];
  const float* bo  = (const float*)d_in[9];
  float* out = (float*)d_out;

  const int nNuc = B_ * N_ * HID;  // 4,194,304
  const int nEle = B_ * M_ * HID;  // 8,388,608

  _Float16* ws    = (_Float16*)d_ws;
  _Float16* nuc16 = ws; ws += nNuc;
  _Float16* ele16 = ws; ws += nEle;
  _Float16* wqt   = ws; ws += 65536;
  _Float16* wkt   = ws; ws += 65536;
  _Float16* wvt   = ws; ws += 65536;
  _Float16* wot   = ws; ws += 65536;
  _Float16* q16   = ws; ws += nNuc;
  _Float16* k16   = ws; ws += nEle;
  _Float16* vt16  = ws; ws += nEle;
  _Float16* o16   = ws; ws += nNuc;

  cvt_f32_to_f16<<<4096, 256, 0, stream>>>(nuc, nuc16, nNuc);
  cvt_f32_to_f16<<<4096, 256, 0, stream>>>(ele, ele16, nEle);

  dim3 tg(16, 16), tb(16, 16);
  transpose256<<<tg, tb, 0, stream>>>(Wq, wqt);
  transpose256<<<tg, tb, 0, stream>>>(Wk, wkt);
  transpose256<<<tg, tb, 0, stream>>>(Wv, wvt);
  transpose256<<<tg, tb, 0, stream>>>(Wo, wot);

  const int rQ = B_ * N_;  // 16384
  const int rE = B_ * M_;  // 32768
  gemm_proj<0><<<(rQ / 16) * 2, 128, 0, stream>>>(nuc16, wqt, bq, q16, N_);
  gemm_proj<0><<<(rE / 16) * 2, 128, 0, stream>>>(ele16, wkt, bk, k16, M_);
  gemm_proj<1><<<(rE / 16) * 2, 128, 0, stream>>>(ele16, wvt, bv, vt16, M_);

  attention_kernel<<<B_ * H_ * (N_ / 64), 128, 0, stream>>>(q16, k16, vt16, o16);

  gemm_proj<2><<<(rQ / 16) * 2, 128, 0, stream>>>(o16, wot, bo, out, N_);
}
